// MultiHeadedSelfAttention_73581379715335
// MI455X (gfx1250) — matrix operand order repaired and checked by reading, not compiled, not run on hardware
//
#include <hip/hip_runtime.h>
#include <hip/hip_bf16.h>
#include <stdint.h>

// Problem constants (match reference)
#define T_TOK 8192
#define NSEG  8
#define SEG   1024
#define NH    16
#define DM    1024
#define HW    64
#define SM_SCALE 0.125f   // 1/sqrt(64)

typedef __bf16 bf16_t;
typedef __attribute__((ext_vector_type(16))) __bf16 v16bf;
typedef __attribute__((ext_vector_type(8)))  __bf16 v8bf;
typedef __attribute__((ext_vector_type(8)))  float  v8f;

union FragU { v16bf v; v8bf h[2]; };

// ---------------------------------------------------------------------------
// Fragment loaders (wave32; A/B use this code's contiguous k order, same for both; C/D is hw layout):
//  A (16x32, MxK): lane = M%16 half=lane/16; element e -> K = half*16 + e
//     => two contiguous 8-elem runs per lane when K is contiguous in memory.
//  B (32x16, KxN): lane col = N, half=lane/16; element e -> K = half*16 + e
//     => 16 contiguous K elements per lane when K is contiguous in memory.
//  C/D (16x16 f32): lane col = N; VGPR r -> M = r + 8*half.
// ---------------------------------------------------------------------------
__device__ __forceinline__ v16bf load_a_frag(const bf16_t* base, int ld, int lane) {
  const int row = lane & 15, half = lane >> 4;
  const bf16_t* p = base + row * ld + half * 16;
  FragU f;
  f.h[0] = *(const v8bf*)(p);        // K = half*16 + 0..7
  f.h[1] = *(const v8bf*)(p + 8);    // K = half*16 + 8..15
  return f.v;
}

__device__ __forceinline__ v16bf load_b_frag(const bf16_t* base, int ld, int lane) {
  const int col = lane & 15, half = lane >> 4;
  return *(const v16bf*)(base + col * ld + half * 16);   // K = half*16 + 0..15
}

__device__ __forceinline__ v8f wmma_bf16(v16bf a, v16bf b, v8f c) {
  return __builtin_amdgcn_wmma_f32_16x16x32_bf16(false, a, false, b, (short)0, c,
                                                 false, false);
}

// ---------------------------------------------------------------------------
// Kernel 1: fp32 -> bf16 convert of x
// ---------------------------------------------------------------------------
__global__ __launch_bounds__(256) void cvt_x_kernel(const float* __restrict__ x,
                                                    bf16_t* __restrict__ xb, int n) {
  for (int i = blockIdx.x * blockDim.x + threadIdx.x; i < n; i += gridDim.x * blockDim.x)
    xb[i] = (bf16_t)x[i];
}

// ---------------------------------------------------------------------------
// Kernel 2: transpose + convert weights: Wt[mat][n][k] = (bf16) W[k][n]
// Tiled 32x32 through LDS so both global sides are coalesced.
// ---------------------------------------------------------------------------
__global__ __launch_bounds__(256) void cvt_wt_kernel(const float* __restrict__ W0,
                                                     const float* __restrict__ W1,
                                                     const float* __restrict__ W2,
                                                     const float* __restrict__ W3,
                                                     bf16_t* __restrict__ wt) {
  __shared__ float tile[32][33];
  const int mat = blockIdx.z;
  const float* Wm = (mat == 0) ? W0 : (mat == 1) ? W1 : (mat == 2) ? W2 : W3;
  bf16_t* dst = wt + (size_t)mat * DM * DM;
  const int nb = blockIdx.x * 32, kb = blockIdx.y * 32;
  const int tx = threadIdx.x, ty = threadIdx.y;     // block (32,8)
#pragma unroll
  for (int i = 0; i < 4; i++)
    tile[ty + 8 * i][tx] = Wm[(size_t)(kb + ty + 8 * i) * DM + nb + tx];
  __syncthreads();
#pragma unroll
  for (int i = 0; i < 4; i++)
    dst[(size_t)(nb + ty + 8 * i) * DM + kb + tx] = (bf16_t)tile[tx][ty + 8 * i];
}

// ---------------------------------------------------------------------------
// Kernel 3: fused QKVC projection GEMM.  y = x @ W + b  for 4 weight sets.
// grid (M/64, N/128, 4), block 128 (4 waves).  Wave strip: 16 rows x 128 cols.
// Software-pipelined: all 8 B fragments land in distinct VGPRs before the
// WMMA burst, and the next A fragment is fetched early, so the scheduler can
// emit one load clause + partial s_wait_loadcnt instead of full round-trips.
// ---------------------------------------------------------------------------
__global__ __launch_bounds__(128) void gemm_qkvc_kernel(
    const bf16_t* __restrict__ xb, const bf16_t* __restrict__ wt,
    const float* __restrict__ bq, const float* __restrict__ bk,
    const float* __restrict__ bv, const float* __restrict__ bc,
    bf16_t* __restrict__ qb, bf16_t* __restrict__ kb2,
    bf16_t* __restrict__ vb, float* __restrict__ cb) {
  const int lane = threadIdx.x & 31;
  const int wave = threadIdx.x >> 5;
  const int m0 = blockIdx.x * 64 + wave * 16;
  const int n0 = blockIdx.y * 128;
  const int mat = blockIdx.z;
  const bf16_t* W = wt + (size_t)mat * DM * DM;   // [N][K], K contiguous
  const bf16_t* arow = xb + (size_t)m0 * DM;

  v8f acc[8] = {};
  v16bf a = load_a_frag(arow, DM, lane);
  for (int k0 = 0; k0 < DM; k0 += 32) {
    v16bf bfr[8];
#pragma unroll
    for (int n = 0; n < 8; n++)
      bfr[n] = load_b_frag(W + (size_t)(n0 + 16 * n) * DM + k0, DM, lane);
    v16bf a_next = a;
    if (k0 + 32 < DM) a_next = load_a_frag(arow + k0 + 32, DM, lane);
#pragma unroll
    for (int n = 0; n < 8; n++)
      acc[n] = wmma_bf16(a, bfr[n], acc[n]);
    a = a_next;
  }

  const float* bias = (mat == 0) ? bq : (mat == 1) ? bk : (mat == 2) ? bv : bc;
  bf16_t* outb = (mat == 0) ? qb : (mat == 1) ? kb2 : vb;
  const int col = lane & 15, half = lane >> 4;
#pragma unroll
  for (int n = 0; n < 8; n++) {
    const int nc = n0 + 16 * n + col;
    const float bval = bias[nc];
#pragma unroll
    for (int r = 0; r < 8; r++) {
      const int row = m0 + r + 8 * half;
      const size_t idx = (size_t)row * DM + nc;
      const float val = acc[n][r] + bval;
      if (mat == 3) cb[idx] = val;
      else          outb[idx] = (bf16_t)val;
    }
  }
}

// ---------------------------------------------------------------------------
// Kernel 4: flash attention per (segment b, head h, 128-row q block) + residual.
// Block = 256 threads = 8 waves; each wave owns a 16-row q strip.
// 64-key K/V tiles staged in LDS (V transposed -> contiguous B fragments).
// Online softmax held entirely in registers (shfl_xor row reductions).
// B fragments for each WMMA burst are batched into arrays so ds-load latency
// overlaps WMMA issue (partial dscnt waits instead of full serialization).
// ---------------------------------------------------------------------------
#define KT_LD 80   // bf16 elements per LDS row (pad; row stride 160B, 32B aligned)
#define PS_LD 72   // probs scratch row stride (144B, 16B aligned)

__global__ __launch_bounds__(256) void attn_kernel(
    const bf16_t* __restrict__ qg, const bf16_t* __restrict__ kg,
    const bf16_t* __restrict__ vg, const float* __restrict__ cg,
    float* __restrict__ out) {
  __shared__ __align__(32) bf16_t kt[64 * KT_LD];          // [key][w]
  __shared__ __align__(32) bf16_t vt[64 * KT_LD];          // [w][key] (transposed)
  __shared__ __align__(32) bf16_t ps[8 * 16 * PS_LD];      // per-wave probs scratch

  const int lane = threadIdx.x & 31;
  const int wave = threadIdx.x >> 5;
  const int h = blockIdx.y;
  const int b = blockIdx.z;
  const int q0 = blockIdx.x * 128 + wave * 16;             // q row within segment
  const size_t tok0 = (size_t)b * SEG;
  const int col = lane & 15, half = lane >> 4;

  // Preload this wave's 16x64 q strip as two A fragments (K = head width).
  v16bf qa[2];
#pragma unroll
  for (int kk = 0; kk < 2; kk++)
    qa[kk] = load_a_frag(qg + (tok0 + q0) * DM + h * HW + kk * 32, DM, lane);

  v8f o[4] = {};
  float m[8], l[8];
#pragma unroll
  for (int r = 0; r < 8; r++) { m[r] = -3.0e38f; l[r] = 0.0f; }

  bf16_t* myps = ps + wave * 16 * PS_LD;

  for (int s0 = 0; s0 < SEG; s0 += 64) {
    __syncthreads();   // protect LDS tiles from previous iteration's readers
    {
      // Cooperative staging: each thread moves one 16-element (32B) chunk.
      const int t = threadIdx.x;
      const int key = t >> 2, ws = (t & 3) * 16;
      v16bf ktmp = *(const v16bf*)(kg + (tok0 + s0 + key) * DM + h * HW + ws);
      *(v16bf*)(kt + key * KT_LD + ws) = ktmp;
      v16bf vtmp = *(const v16bf*)(vg + (tok0 + s0 + key) * DM + h * HW + ws);
#pragma unroll
      for (int e = 0; e < 16; e++)
        vt[(ws + e) * KT_LD + key] = vtmp[e];              // transpose into LDS
    }
    __syncthreads();

    // scores: 16 q rows x 64 keys = 4 C tiles, contraction over w (two K steps)
    v16bf kbf[8];
#pragma unroll
    for (int t = 0; t < 4; t++)
#pragma unroll
      for (int kk = 0; kk < 2; kk++)
        kbf[2 * t + kk] =
            *(const v16bf*)(kt + (16 * t + col) * KT_LD + kk * 32 + half * 16);
    v8f sc[4] = {};
#pragma unroll
    for (int t = 0; t < 4; t++)
#pragma unroll
      for (int kk = 0; kk < 2; kk++)
        sc[t] = wmma_bf16(qa[kk], kbf[2 * t + kk], sc[t]);

    // Online softmax, rows distributed as M = r + 8*half per C VGPR r.
#pragma unroll
    for (int r = 0; r < 8; r++) {
      float tm = -3.0e38f;
#pragma unroll
      for (int t = 0; t < 4; t++) {
        sc[t][r] *= SM_SCALE;
        tm = fmaxf(tm, sc[t][r]);
      }
      tm = fmaxf(tm, __shfl_xor(tm, 1, 32));
      tm = fmaxf(tm, __shfl_xor(tm, 2, 32));
      tm = fmaxf(tm, __shfl_xor(tm, 4, 32));
      tm = fmaxf(tm, __shfl_xor(tm, 8, 32));
      const float mn = fmaxf(m[r], tm);
      const float corr = __expf(m[r] - mn);
      m[r] = mn;
      l[r] *= corr;
#pragma unroll
      for (int nt = 0; nt < 4; nt++) o[nt][r] *= corr;
      float rs = 0.0f;
#pragma unroll
      for (int t = 0; t < 4; t++) {
        const float p = __expf(sc[t][r] - mn);
        sc[t][r] = p;
        rs += p;
      }
      rs += __shfl_xor(rs, 1, 32);
      rs += __shfl_xor(rs, 2, 32);
      rs += __shfl_xor(rs, 4, 32);
      rs += __shfl_xor(rs, 8, 32);
      l[r] += rs;
    }

    // C-layout -> A-layout via per-wave LDS scratch (bf16 probs).
#pragma unroll
    for (int t = 0; t < 4; t++)
#pragma unroll
      for (int r = 0; r < 8; r++)
        myps[(r + 8 * half) * PS_LD + 16 * t + col] = (bf16_t)sc[t][r];
    asm volatile("s_wait_dscnt 0" ::: "memory");   // wave-local LDS RAW fence

    v16bf pa[2];
#pragma unroll
    for (int kk = 0; kk < 2; kk++)
      pa[kk] = load_a_frag(myps + kk * 32, PS_LD, lane);

    // out += probs @ v ; B fragments contiguous from transposed vt tile.
    v16bf vbf[8];
#pragma unroll
    for (int nt = 0; nt < 4; nt++)
#pragma unroll
      for (int kk = 0; kk < 2; kk++)
        vbf[2 * nt + kk] =
            *(const v16bf*)(vt + (16 * nt + col) * KT_LD + kk * 32 + half * 16);
#pragma unroll
    for (int nt = 0; nt < 4; nt++)
#pragma unroll
      for (int kk = 0; kk < 2; kk++)
        o[nt] = wmma_bf16(pa[kk], vbf[2 * nt + kk], o[nt]);
  }

  // Normalize, add residual c, store fp32.
#pragma unroll
  for (int nt = 0; nt < 4; nt++) {
#pragma unroll
    for (int r = 0; r < 8; r++) {
      const int row = q0 + r + 8 * half;
      const size_t idx = (tok0 + row) * DM + h * HW + 16 * nt + col;
      out[idx] = o[nt][r] / l[r] + cg[idx];
    }
  }
}

// ---------------------------------------------------------------------------
// Host launcher. Workspace layout (all offsets 256B aligned), ~104 MB total:
//   [0,16M)      x bf16
//   [16M,24M)    Wt bf16 x4 (transposed weights, [N][K])
//   [24M..]      q,k,v bf16 (16MB each), c fp32 (32MB)
// ---------------------------------------------------------------------------
extern "C" void kernel_launch(void* const* d_in, const int* in_sizes, int n_in,
                              void* d_out, int out_size, void* d_ws, size_t ws_size,
                              hipStream_t stream) {
  (void)in_sizes; (void)n_in; (void)out_size; (void)ws_size;
  const float* x  = (const float*)d_in[0];
  // d_in[1] = splits (equal segments; layout is fixed, unused)
  const float* Wq = (const float*)d_in[2];
  const float* bq = (const float*)d_in[3];
  const float* Wk = (const float*)d_in[4];
  const float* bk = (const float*)d_in[5];
  const float* Wv = (const float*)d_in[6];
  const float* bv = (const float*)d_in[7];
  const float* Wc = (const float*)d_in[8];
  const float* bc = (const float*)d_in[9];
  float* out = (float*)d_out;

  char* ws = (char*)d_ws;
  const size_t XB_OFF = 0;
  const size_t WT_OFF = (size_t)T_TOK * DM * 2;                 // 16 MB
  const size_t QB_OFF = WT_OFF + (size_t)4 * DM * DM * 2;       // +8 MB
  const size_t KB_OFF = QB_OFF + (size_t)T_TOK * DM * 2;
  const size_t VB_OFF = KB_OFF + (size_t)T_TOK * DM * 2;
  const size_t CB_OFF = VB_OFF + (size_t)T_TOK * DM * 2;

  bf16_t* xb = (bf16_t*)(ws + XB_OFF);
  bf16_t* wt = (bf16_t*)(ws + WT_OFF);
  bf16_t* qb = (bf16_t*)(ws + QB_OFF);
  bf16_t* kb2 = (bf16_t*)(ws + KB_OFF);
  bf16_t* vb = (bf16_t*)(ws + VB_OFF);
  float*  cb = (float*)(ws + CB_OFF);

  const int n_x = T_TOK * DM;
  cvt_x_kernel<<<4096, 256, 0, stream>>>(x, xb, n_x);
  cvt_wt_kernel<<<dim3(DM / 32, DM / 32, 4), dim3(32, 8), 0, stream>>>(Wq, Wk, Wv, Wc, wt);
  gemm_qkvc_kernel<<<dim3(T_TOK / 64, DM / 128, 4), 128, 0, stream>>>(
      xb, wt, bq, bk, bv, bc, qb, kb2, vb, cb);
  attn_kernel<<<dim3(SEG / 128, NH, NSEG), 256, 0, stream>>>(qb, kb2, vb, cb, out);
}
